// PointNetSetAbstraction_91328184582663
// MI455X (gfx1250) — matrix operand order repaired and checked by reading, not compiled, not run on hardware
//
#include <hip/hip_runtime.h>

#define B_  16
#define N_  4096
#define S_  1024
#define NS_ 32
#define M_  (B_*S_*NS_)   // 524288 rows

typedef _Float16 f16;
typedef __attribute__((ext_vector_type(16))) _Float16 v16h;
typedef __attribute__((ext_vector_type(8)))  _Float16 v8h;
typedef __attribute__((ext_vector_type(8)))  float    v8f;

// ---------------- WMMA fragment loaders (CDNA5 16x16x32 f16 layout) ----------
// A (16x32, MxK): lane L holds row M=L%16. Lanes 0-15: elems 0-7 -> K 0-7,
// elems 8-15 -> K 16-23. Lanes 16-31: elems 0-7 -> K 8-15, elems 8-15 -> K 24-31.
__device__ __forceinline__ v16h load_a_frag(const f16* __restrict__ A, int lda,
                                            int m0, int kbase, int lane) {
  int row = m0 + (lane & 15);
  int kb  = kbase + ((lane >> 4) << 3);          // +8 halves for hi lanes
  const f16* p = A + (size_t)row * lda + kb;
  v8h lo = *(const v8h*)p;                       // K = kb .. kb+7
  v8h hi = *(const v8h*)(p + 16);                // K = kb+16 .. kb+23
  return __builtin_shufflevector(lo, hi, 0,1,2,3,4,5,6,7,8,9,10,11,12,13,14,15);
}

// B (32x16, KxN) is pre-swizzled by prep_w_kernel so each lane reads 16
// contiguous halves: wp[frag*512 + lane*16 + e], e -> K = (e&7) + (laneHi?8:0) + (e>=8?16:0).
__device__ __forceinline__ v16h load_b_frag(const f16* __restrict__ wp,
                                            int frag, int lane) {
  return *(const v16h*)(wp + (size_t)frag * 512 + lane * 16);
}

// ---------------- weight prep: f32 (O,C) -> f16 B-fragment order -------------
// segments (halves): L1 [0,2048) NT=4,KC=1,K=9 ; L2 [2048,6144) NT=4,KC=2,K=64 ;
// L3 [6144,14336) NT=8,KC=2,K=64
__global__ void prep_w_kernel(const float* __restrict__ w0,
                              const float* __restrict__ w1,
                              const float* __restrict__ w2,
                              f16* __restrict__ wp) {
  int t = blockIdx.x * blockDim.x + threadIdx.x;
  if (t >= 14336) return;
  const float* w; int K, KC, rel;
  if (t < 2048)      { w = w0; K = 9;  KC = 1; rel = t; }
  else if (t < 6144) { w = w1; K = 64; KC = 2; rel = t - 2048; }
  else               { w = w2; K = 64; KC = 2; rel = t - 6144; }
  int e    = rel & 15;
  int lane = (rel >> 4) & 31;
  int frag = rel >> 9;                 // frag = nt*KC + kc
  int kc   = frag % KC;
  int nt   = frag / KC;
  int n = nt * 16 + (lane & 15);
  int k = kc * 32 + (e & 7) + ((lane >= 16) ? 8 : 0) + ((e >= 8) ? 16 : 0);
  wp[t] = (k < K) ? (f16)w[n * K + k] : (f16)0.f;
}

// ---------------- FPS: one block per batch, serial over S ---------------------
__global__ __launch_bounds__(256)
void fps_kernel(const float* __restrict__ xyz, int* __restrict__ fps_idx) {
  int b = blockIdx.x, t = threadIdx.x;
  const float* X = xyz + (size_t)b * 3 * N_;     // layout (3, N)
  float px[16], py[16], pz[16], dd[16];
#pragma unroll
  for (int i = 0; i < 16; i++) {
    int n = t + i * 256;
    px[i] = X[n]; py[i] = X[N_ + n]; pz[i] = X[2 * N_ + n];
    dd[i] = 1e10f;
  }
  __shared__ float rv[256];
  __shared__ int   ri[256];
  __shared__ float cc[3];
  __shared__ int   cur;
  if (t == 0) cur = 0;
  __syncthreads();
  for (int it = 0; it < S_; it++) {
    if (t == 0) {
      fps_idx[b * S_ + it] = cur;
      cc[0] = X[cur]; cc[1] = X[N_ + cur]; cc[2] = X[2 * N_ + cur];
    }
    __syncthreads();
    float cx = cc[0], cy = cc[1], cz = cc[2];
    float bv = -1.f; int bi = 0;
#pragma unroll
    for (int i = 0; i < 16; i++) {
      float dx = px[i] - cx, dy = py[i] - cy, dz = pz[i] - cz;
      float d = dx * dx + dy * dy + dz * dz;
      d = fminf(d, dd[i]); dd[i] = d;
      int n = t + i * 256;
      if (d > bv || (d == bv && n < bi)) { bv = d; bi = n; }
    }
    rv[t] = bv; ri[t] = bi;
    __syncthreads();
    for (int st = 128; st > 0; st >>= 1) {
      if (t < st) {
        float ov = rv[t + st]; int oi = ri[t + st];
        if (ov > rv[t] || (ov == rv[t] && oi < ri[t])) { rv[t] = ov; ri[t] = oi; }
      }
      __syncthreads();
    }
    if (t == 0) cur = ri[0];
    __syncthreads();
  }
}

// ---------------- ball query + grouping: one wave32 per (b,s) -----------------
__global__ __launch_bounds__(256)
void group_kernel(const float* __restrict__ xyz, const float* __restrict__ pts,
                  const int* __restrict__ fps_idx, f16* __restrict__ A0,
                  float* __restrict__ out_xyz) {
  __shared__ int sIdx[8][32];
  int t = threadIdx.x, lane = t & 31, wid = t >> 5;
  int g = blockIdx.x * 8 + wid;                  // 0 .. B*S-1
  int b = g >> 10, s = g & (S_ - 1);
  const float* X = xyz + (size_t)b * 3 * N_;
  const float* P = pts + (size_t)b * 6 * N_;
  int fi = fps_idx[g];
  float cx = X[fi], cy = X[N_ + fi], cz = X[2 * N_ + fi];
  if (lane == 0) {
    out_xyz[((size_t)b * 3 + 0) * S_ + s] = cx;
    out_xyz[((size_t)b * 3 + 1) * S_ + s] = cy;
    out_xyz[((size_t)b * 3 + 2) * S_ + s] = cz;
  }
  const float r2 = 0.4f * 0.4f;
  int cnt = 0, first = -1;
  for (int base = 0; base < N_; base += 32) {
    int n = base + lane;
    float dx = X[n] - cx, dy = X[N_ + n] - cy, dz = X[2 * N_ + n] - cz;
    bool pred = (dx * dx + dy * dy + dz * dz) <= r2;
    unsigned bal = __builtin_amdgcn_ballot_w32(pred);
    if (first < 0 && bal) first = base + __builtin_ctz(bal);
    int off = __popc(bal & ((1u << lane) - 1u));
    int pos = cnt + off;
    if (pred && pos < 32) sIdx[wid][pos] = n;
    cnt += __popc(bal);
    if (cnt >= 32) break;                        // wave-uniform
  }
  if (cnt > 32) cnt = 32;
  if (lane >= cnt) sIdx[wid][lane] = first;      // pad with first idx (ref semantics)
  __syncthreads();
  int id = sIdx[wid][lane];
  f16 row[32];
#pragma unroll
  for (int i = 0; i < 32; i++) row[i] = (f16)0.f;
  row[0] = (f16)(X[id] - cx);
  row[1] = (f16)(X[N_ + id] - cy);
  row[2] = (f16)(X[2 * N_ + id] - cz);
#pragma unroll
  for (int c = 0; c < 6; c++) row[3 + c] = (f16)P[c * N_ + id];
  uint4* dst = (uint4*)(A0 + ((size_t)g * 32 + lane) * 32);
  const uint4* src = (const uint4*)row;
  dst[0] = src[0]; dst[1] = src[1]; dst[2] = src[2]; dst[3] = src[3];
}

// ---------------- GEMM pass 1: per-channel sum / sumsq ------------------------
template <int KC, int NT>
__global__ __launch_bounds__(256)
void gemm_stats_kernel(const f16* __restrict__ A, const f16* __restrict__ wp,
                       float* __restrict__ stats) {
  __shared__ float ssum[NT * 16], ssq[NT * 16];
  int t = threadIdx.x;
  if (t < NT * 16) { ssum[t] = 0.f; ssq[t] = 0.f; }
  __syncthreads();
  int lane = t & 31, wid = t >> 5;
  int m0 = (blockIdx.x * 8 + wid) * 16;
  const int lda = KC * 32;
  v16h a[KC];
#pragma unroll
  for (int kc = 0; kc < KC; kc++) a[kc] = load_a_frag(A, lda, m0, kc * 32, lane);
#pragma unroll
  for (int nt = 0; nt < NT; nt++) {
    v8f c = {};
#pragma unroll
    for (int kc = 0; kc < KC; kc++) {
      v16h bf = load_b_frag(wp, nt * KC + kc, lane);
      c = __builtin_amdgcn_wmma_f32_16x16x32_f16(false, a[kc], false, bf,
                                                 (short)0, c, false, false);
    }
    float s1 = 0.f, s2 = 0.f;
#pragma unroll
    for (int v = 0; v < 8; v++) { s1 += c[v]; s2 += c[v] * c[v]; }
    s1 += __shfl_xor(s1, 16, 32);
    s2 += __shfl_xor(s2, 16, 32);
    if (lane < 16) {
      atomicAdd(&ssum[nt * 16 + lane], s1);
      atomicAdd(&ssq[nt * 16 + lane], s2);
    }
  }
  __syncthreads();
  if (t < NT * 16) {
    atomicAdd(&stats[t], ssum[t]);
    atomicAdd(&stats[NT * 16 + t], ssq[t]);
  }
}

// ---------------- stats -> (scale, shift) ------------------------------------
__global__ void finalize_kernel(const float* __restrict__ stats,
                                const float* __restrict__ g,
                                const float* __restrict__ beta,
                                float* __restrict__ scsh, int C, float invM) {
  int n = threadIdx.x;
  if (n >= C) return;
  float mu  = stats[n] * invM;
  float var = stats[C + n] * invM - mu * mu;
  float sc  = g[n] * rsqrtf(var + 1e-5f);
  scsh[n]     = sc;
  scsh[C + n] = beta[n] - mu * sc;
}

// ---------------- GEMM pass 2: BN + ReLU -> f16 next-layer input --------------
template <int KC, int NT>
__global__ __launch_bounds__(256)
void gemm_bnstore_kernel(const f16* __restrict__ A, const f16* __restrict__ wp,
                         const float* __restrict__ scsh, f16* __restrict__ Anext) {
  int t = threadIdx.x, lane = t & 31, wid = t >> 5;
  int m0 = (blockIdx.x * 8 + wid) * 16;
  const int lda = KC * 32;
  const int Cout = NT * 16;
  v16h a[KC];
#pragma unroll
  for (int kc = 0; kc < KC; kc++) a[kc] = load_a_frag(A, lda, m0, kc * 32, lane);
#pragma unroll
  for (int nt = 0; nt < NT; nt++) {
    v8f c = {};
#pragma unroll
    for (int kc = 0; kc < KC; kc++) {
      v16h bf = load_b_frag(wp, nt * KC + kc, lane);
      c = __builtin_amdgcn_wmma_f32_16x16x32_f16(false, a[kc], false, bf,
                                                 (short)0, c, false, false);
    }
    int n = nt * 16 + (lane & 15);
    float sc = scsh[n], sh = scsh[Cout + n];
    int rbase = m0 + ((lane >> 4) << 3);
#pragma unroll
    for (int v = 0; v < 8; v++) {
      float d = c[v] * sc + sh;
      d = d > 0.f ? d : 0.f;
      Anext[(size_t)(rbase + v) * Cout + n] = (f16)d;
    }
  }
}

// ---------------- layer 3 pass 2: BN + ReLU + max over ns fused ---------------
__global__ __launch_bounds__(256)
void gemm_maxpool_kernel(const f16* __restrict__ A, const f16* __restrict__ wp,
                         const float* __restrict__ scsh, float* __restrict__ out) {
  int t = threadIdx.x, lane = t & 31, wid = t >> 5;
  int g = blockIdx.x * 8 + wid;                  // (b,s) group
  int b = g >> 10, s = g & (S_ - 1);
  int m0 = g * 32;                               // ns=32 rows = two M-tiles
  v16h a0[2], a1[2];
#pragma unroll
  for (int kc = 0; kc < 2; kc++) {
    a0[kc] = load_a_frag(A, 64, m0,      kc * 32, lane);
    a1[kc] = load_a_frag(A, 64, m0 + 16, kc * 32, lane);
  }
#pragma unroll
  for (int nt = 0; nt < 8; nt++) {
    v8f c0 = {}, c1 = {};
#pragma unroll
    for (int kc = 0; kc < 2; kc++) {
      v16h bf = load_b_frag(wp, nt * 2 + kc, lane);
      c0 = __builtin_amdgcn_wmma_f32_16x16x32_f16(false, a0[kc], false, bf,
                                                  (short)0, c0, false, false);
      c1 = __builtin_amdgcn_wmma_f32_16x16x32_f16(false, a1[kc], false, bf,
                                                  (short)0, c1, false, false);
    }
    int n = nt * 16 + (lane & 15);
    float sc = scsh[n], sh = scsh[128 + n];
    float mx = 0.f;                              // max(relu(x)) == max(0, max x)
#pragma unroll
    for (int v = 0; v < 8; v++) {
      mx = fmaxf(mx, c0[v] * sc + sh);
      mx = fmaxf(mx, c1[v] * sc + sh);
    }
    mx = fmaxf(mx, __shfl_xor(mx, 16, 32));
    if (lane < 16)
      out[((size_t)b * 128 + nt * 16 + lane) * S_ + s] = mx;
  }
}

__global__ void zero_kernel(float* p, int n) {
  int t = blockIdx.x * blockDim.x + threadIdx.x;
  if (t < n) p[t] = 0.f;
}

// -----------------------------------------------------------------------------
extern "C" void kernel_launch(void* const* d_in, const int* in_sizes, int n_in,
                              void* d_out, int out_size, void* d_ws, size_t ws_size,
                              hipStream_t stream) {
  (void)in_sizes; (void)n_in; (void)out_size; (void)ws_size;
  const float* xyz = (const float*)d_in[0];
  const float* pts = (const float*)d_in[1];
  const float* w0  = (const float*)d_in[2];
  const float* g0  = (const float*)d_in[4];
  const float* be0 = (const float*)d_in[5];
  const float* w1  = (const float*)d_in[6];
  const float* g1  = (const float*)d_in[8];
  const float* be1 = (const float*)d_in[9];
  const float* w2  = (const float*)d_in[10];
  const float* g2  = (const float*)d_in[12];
  const float* be2 = (const float*)d_in[13];

  char* ws = (char*)d_ws;
  size_t off = 0;
  auto alloc = [&](size_t bytes) {
    size_t o = off;
    off = (off + bytes + 255) & ~(size_t)255;
    return o;
  };
  int*   fps_idx = (int*)  (ws + alloc((size_t)B_ * S_ * 4));
  float* stats   = (float*)(ws + alloc(3 * 256 * 4));
  float* scsh    = (float*)(ws + alloc(3 * 256 * 4));
  f16*   wp      = (f16*)  (ws + alloc(14336 * 2));
  f16*   A0      = (f16*)  (ws + alloc((size_t)M_ * 32 * 2));   // 32 MB
  f16*   A1      = (f16*)  (ws + alloc((size_t)M_ * 64 * 2));   // 64 MB
  f16*   A2      = (f16*)  (ws + alloc((size_t)M_ * 64 * 2));   // 64 MB

  float* out_xyz = (float*)d_out;
  float* out_pts = (float*)d_out + (size_t)B_ * 3 * S_;

  const int gemm_blocks = M_ / 16 / 8;  // 4096
  const float invM = 1.f / (float)M_;

  zero_kernel<<<1, 768, 0, stream>>>(stats, 768);
  prep_w_kernel<<<(14336 + 255) / 256, 256, 0, stream>>>(w0, w1, w2, wp);
  fps_kernel<<<B_, 256, 0, stream>>>(xyz, fps_idx);
  group_kernel<<<(B_ * S_) / 8, 256, 0, stream>>>(xyz, pts, fps_idx, A0, out_xyz);

  // layer 1: K=9 (pad 32), Cout=64
  gemm_stats_kernel<1, 4><<<gemm_blocks, 256, 0, stream>>>(A0, wp, stats);
  finalize_kernel<<<1, 128, 0, stream>>>(stats, g0, be0, scsh, 64, invM);
  gemm_bnstore_kernel<1, 4><<<gemm_blocks, 256, 0, stream>>>(A0, wp, scsh, A1);

  // layer 2: K=64, Cout=64
  gemm_stats_kernel<2, 4><<<gemm_blocks, 256, 0, stream>>>(A1, wp + 2048, stats + 256);
  finalize_kernel<<<1, 128, 0, stream>>>(stats + 256, g1, be1, scsh + 256, 64, invM);
  gemm_bnstore_kernel<2, 4><<<gemm_blocks, 256, 0, stream>>>(A1, wp + 2048, scsh + 256, A2);

  // layer 3: K=64, Cout=128, fused BN+ReLU+maxpool
  gemm_stats_kernel<2, 8><<<gemm_blocks, 256, 0, stream>>>(A2, wp + 6144, stats + 512);
  finalize_kernel<<<1, 128, 0, stream>>>(stats + 512, g2, be2, scsh + 512, 128, invM);
  gemm_maxpool_kernel<<<(B_ * S_) / 8, 256, 0, stream>>>(A2, wp + 6144, scsh + 512, out_pts);
}
